// BoundaryGuidedUpsample_51926154609228
// MI455X (gfx1250) — matrix operand order repaired and checked by reading, not compiled, not run on hardware
//
#include <hip/hip_runtime.h>
#include <math.h>

// ---------------------------------------------------------------------------
// BoundaryGuidedUpsample for MI455X (gfx1250, wave32, WMMA).
//
// B=2, H0=W0=64, R=4, WIN=3 (K=9), C=22 (l4:9 + l6:13); 131072 HR pixels.
// Both tensor-product contractions run as WMMA f16 GEMMs against static
// matrices built on device each launch (deterministic):
//   Gmat[rpos=16][kstep=7][ntile=2][lane=32][16 halfs]  (f16, WMMA-B layout)
//   CGo2[ntile=31][lane=32][16 halfs]                   (f16, WMMA-B layout)
// Real-basis Wigner-3j computed on device in double (Racah + r2c basis).
//
// Main kernel: 64 threads (2 wave32) per 4x4 LR block x 4 subpixel phases.
// The attention-scaled window vector z (224 f16, zero padded) is staged in
// LDS so each WMMA A operand is two ds_load_b128 + a register shuffle.
// ---------------------------------------------------------------------------

typedef __attribute__((ext_vector_type(16))) _Float16 v16h;
typedef __attribute__((ext_vector_type(8)))  _Float16 v8h;
typedef __attribute__((ext_vector_type(8)))  float    v8f;

#define DEV __device__ __forceinline__

// Path tables: 0..5 = PATHS_VAL, 6..13 = PATHS_OUT
__device__ const int PL1[14]  = {4,4,6,4,6,6, 4,4,6,6,4,4,6,6};
__device__ const int PL2[14]  = {0,2,2,2,0,2, 4,6,4,6,4,6,4,6};
__device__ const int PL3[14]  = {4,4,4,6,6,6, 4,4,4,4,6,6,6,6};
__device__ const int POFF[14] = {0,81,486,1071,1656,1825,2670,3399,4452,5505,7026,8079,9600,11121};

// --- workspace byte offsets (all 256-aligned); total ~708 KB ---
#define WS_CG    0          // 13318 f32
#define WS_GMAT  53504      // 16*7*2*32*16 = 114688 f16
#define WS_CGO2  512256     // 31*32*16    =  15872 f16
#define WS_INV   544000     // 2*4096*2 f32
#define WS_BH    609536     // 2*4096 f32
#define WS_BV    642304     // 2*4096 f32
#define WS_BM    675072     // 2*4096 f32

DEV int iclampi(int v, int lo, int hi) { return v < lo ? lo : (v > hi ? hi : v); }

DEV unsigned pkh(float a, float b) {   // pack 2 x f32 -> 2 x f16 (RNE)
    union { _Float16 h[2]; unsigned u; } x;
    x.h[0] = (_Float16)a; x.h[1] = (_Float16)b;
    return x.u;
}

// ------------------------- complex double helpers --------------------------
struct cplx { double re, im; };
DEV cplx cmul(cplx a, cplx b) { return {a.re*b.re - a.im*b.im, a.re*b.im + a.im*b.re}; }
DEV cplx cconj(cplx a) { return {a.re, -a.im}; }
DEV cplx cadd(cplx a, cplx b) { return {a.re + b.re, a.im + b.im}; }
DEV cplx cscale(cplx a, double s) { return {a.re*s, a.im*s}; }

DEV double fct(int n) { double f = 1.0; for (int i = 2; i <= n; ++i) f *= (double)i; return f; }

DEV double su2cg(int j1, int j2, int j3, int m1, int m2, int m3) {
    if (m1 + m2 != m3) return 0.0;
    int am3 = m3 < 0 ? -m3 : m3;
    if (am3 > j3) return 0.0;
    double pref = sqrt((2.0*j3 + 1.0) * fct(j3+j1-j2) * fct(j3-j1+j2) * fct(j1+j2-j3) / fct(j1+j2+j3+1));
    pref *= sqrt(fct(j3+m3)*fct(j3-m3)*fct(j1-m1)*fct(j1+m1)*fct(j2-m2)*fct(j2+m2));
    double s = 0.0;
    for (int k = 0; k <= j1 + j2 - j3; ++k) {
        int d1 = j1+j2-j3-k, d2 = j1-m1-k, d3 = j2+m2-k, d4 = j3-j2+m1+k, d5 = j3-j1-m2+k;
        if (d1 < 0 || d2 < 0 || d3 < 0 || d4 < 0 || d5 < 0) continue;
        double den = fct(k)*fct(d1)*fct(d2)*fct(d3)*fct(d4)*fct(d5);
        s += ((k & 1) ? -1.0 : 1.0) / den;
    }
    return pref * s;
}

DEV cplx qent(int l, int a, int i) {   // q_r2c entry sans (-i)^l prefactor
    const double iv = 0.70710678118654752440;
    int m = a - l;
    cplx z = {0.0, 0.0};
    if (m == 0) { if (i == l) z.re = 1.0; }
    else if (m < 0) {
        int am = -m;
        if (i == l + am) z.re = iv;
        else if (i == l - am) z.im = -iv;
    } else {
        double sg = (m & 1) ? -1.0 : 1.0;
        if (i == l + m) z.re = sg * iv;
        else if (i == l - m) z.im = sg * iv;
    }
    return z;
}

DEV cplx miPow(int l) {  // (-i)^l
    switch (l & 3) {
        case 0: return {1.0, 0.0};
        case 1: return {0.0, -1.0};
        case 2: return {-1.0, 0.0};
        default: return {0.0, 1.0};
    }
}

// ------------------- setup: real-basis Wigner 3j tensors -------------------
__global__ void k_cg(float* __restrict__ cg) {
    int p = threadIdx.x;
    if (p >= 14) return;
    const int l1 = PL1[p], l2 = PL2[p], l3 = PL3[p];
    const int n1 = 2*l1+1, n2 = 2*l2+1, n3 = 2*l3+1;
    float* out = cg + POFF[p];
    cplx ph = cmul(cmul(miPow(l1), miPow(l2)), cconj(miPow(l3)));
    double nrm2 = 0.0;
    for (int i = 0; i < n1; ++i) {
        int d1 = i > l1 ? i - l1 : l1 - i;
        for (int j = 0; j < n2; ++j) {
            int d2 = j > l2 ? j - l2 : l2 - j;
            for (int k = 0; k < n3; ++k) {
                int d3 = k > l3 ? k - l3 : l3 - k;
                cplx acc = {0.0, 0.0};
                for (int ia = 0; ia < (d1 ? 2 : 1); ++ia) {
                    int a = l1 + (ia ? d1 : -d1);
                    cplx q1 = qent(l1, a, i);
                    for (int ib = 0; ib < (d2 ? 2 : 1); ++ib) {
                        int bq = l2 + (ib ? d2 : -d2);
                        cplx q12 = cmul(q1, qent(l2, bq, j));
                        for (int ic = 0; ic < (d3 ? 2 : 1); ++ic) {
                            int cq = l3 + (ic ? d3 : -d3);
                            double w3 = su2cg(l1, l2, l3, a - l1, bq - l2, cq - l3);
                            if (w3 == 0.0) continue;
                            acc = cadd(acc, cscale(cmul(q12, cconj(qent(l3, cq, k))), w3));
                        }
                    }
                }
                double v = ph.re * acc.re - ph.im * acc.im;
                out[(i*n2 + j)*n3 + k] = (float)v;
                nrm2 += v * v;
            }
        }
    }
    float inv = (float)(1.0 / sqrt(nrm2));
    for (int t = 0; t < n1*n2*n3; ++t) out[t] *= inv;
}

// spherical harmonics (l=0 and l=2 block), e3nn y-up convention
DEV void sh06(int rpos, int kw, float* sh) {
    float ryf = (float)(rpos >> 2) * 0.25f;
    float rxf = (float)(rpos & 3) * 0.25f;
    float oy = (float)(kw / 3 - 1);
    float ox = (float)(kw % 3 - 1);
    float dy = ryf - oy, dx = rxf - ox;
    float r2 = dx*dx + dy*dy;
    float X, Y, Z;
    if (r2 < 1e-8f) { X = 0.f; Y = 0.f; Z = 1.f; }
    else { float ir = 1.f / sqrtf(r2); X = dx * ir; Y = dy * ir; Z = 0.f; }
    const float C1f = 1.09254843059207918f;
    const float C0f = 0.31539156525252005f;
    const float C2f = 0.54627421529603959f;
    const float Y00 = 0.28209479177387814f;
    sh[0] = Y00;
    sh[1] = C1f * X * Z;
    sh[2] = C1f * X * Y;
    sh[3] = C0f * (3.f*Y*Y - 1.f);
    sh[4] = C1f * Y * Z;
    sh[5] = C2f * (Z*Z - X*X);
}

// ------- setup: context GEMM matrices, emitted in WMMA-B lane layout -------
__global__ void k_gmat(const float* __restrict__ cg, const float* __restrict__ wval,
                       _Float16* __restrict__ gm) {
    int idx = blockIdx.x * blockDim.x + threadIdx.x;
    if (idx >= 16*7*2*32*16) return;
    int h = idx & 15;  int t1 = idx >> 4;
    int lane = t1 & 31; t1 >>= 5;
    int nt = t1 & 1;    t1 >>= 1;
    int ks = t1 % 7;    int rpos = t1 / 7;
    int kk = h + ((lane & 16) ? 16 : 0);      // B: lanes 16-31 hold K=16..31
    int c3 = nt * 16 + (lane & 15);           // B: column N = lane%16
    int Kg = ks * 32 + kk;
    float val = 0.f;
    if (c3 < 22 && Kg < 198) {
        int kw = Kg / 22, c1 = Kg - kw * 22;
        float sh[6]; sh06(rpos, kw, sh);
        int l1 = (c1 < 9) ? 4 : 6; int i = (c1 < 9) ? c1 : c1 - 9;
        int l3 = (c3 < 9) ? 4 : 6; int k3 = (c3 < 9) ? c3 : c3 - 9;
        const float a4 = 1.73205080756887729f;  // sqrt(9)/sqrt(3)
        const float a6 = 2.08166599946613273f;  // sqrt(13)/sqrt(3)
        if (l1 == 4 && l3 == 4) {
            val += wval[0] * a4 * cg[POFF[0] + i*9 + k3] * sh[0];
            for (int j = 0; j < 5; ++j) val += wval[1] * a4 * cg[POFF[1] + (i*5 + j)*9 + k3] * sh[1+j];
        } else if (l1 == 6 && l3 == 4) {
            for (int j = 0; j < 5; ++j) val += wval[2] * a4 * cg[POFF[2] + (i*5 + j)*9 + k3] * sh[1+j];
        } else if (l1 == 4 && l3 == 6) {
            for (int j = 0; j < 5; ++j) val += wval[3] * a6 * cg[POFF[3] + (i*5 + j)*13 + k3] * sh[1+j];
        } else {
            val += wval[4] * a6 * cg[POFF[4] + i*13 + k3] * sh[0];
            for (int j = 0; j < 5; ++j) val += wval[5] * a6 * cg[POFF[5] + (i*5 + j)*13 + k3] * sh[1+j];
        }
    }
    gm[idx] = (_Float16)val;
}

// ---- setup: output TP matrix CGo[c1][c2][c3] as 22x484, WMMA-B layout ----
__global__ void k_cgo2(const float* __restrict__ cg, const float* __restrict__ wout,
                       _Float16* __restrict__ go) {
    int idx = blockIdx.x * blockDim.x + threadIdx.x;
    if (idx >= 31*32*16) return;
    int h = idx & 15; int t1 = idx >> 4;
    int lane = t1 & 31; int q16 = t1 >> 5;   // 0..30
    int c2 = h + ((lane & 16) ? 16 : 0);
    int col = q16 * 16 + (lane & 15);        // col = c1*22 + c3
    float val = 0.f;
    if (c2 < 22 && col < 484) {
        int c1 = col / 22, c3 = col - c1 * 22;
        int l1 = (c1 < 9) ? 4 : 6, i  = (c1 < 9) ? c1 : c1 - 9;
        int l2 = (c2 < 9) ? 4 : 6, j  = (c2 < 9) ? c2 : c2 - 9;
        int l3 = (c3 < 9) ? 4 : 6, k3 = (c3 < 9) ? c3 : c3 - 9;
        int p  = ((l3 == 6) ? 4 : 0) + ((l1 == 6) ? 2 : 0) + ((l2 == 6) ? 1 : 0);
        int pi = 6 + p;
        int n2 = 2*PL2[pi] + 1, n3 = 2*PL3[pi] + 1;
        float alpha = (l3 == 4) ? 1.5f : 1.80277563773199465f;  // sqrt(2l3+1)/2
        val = wout[p] * alpha * cg[POFF[pi] + (i*n2 + j)*n3 + k3];
    }
    go[idx] = (_Float16)val;
}

// ----------------------------- LR prep kernels -----------------------------
__global__ void k_norm(const float* __restrict__ f4, const float* __restrict__ f6,
                       float* __restrict__ invn) {
    int i = blockIdx.x * blockDim.x + threadIdx.x;
    if (i >= 8192) return;
    float s4 = 0.f, s6 = 0.f;
    #pragma unroll
    for (int c = 0; c < 9; ++c)  { float v = f4[i*9 + c];  s4 += v*v; }
    #pragma unroll
    for (int c = 0; c < 13; ++c) { float v = f6[i*13 + c]; s6 += v*v; }
    invn[i*2 + 0] = 1.f / fmaxf(sqrtf(s4), 1e-12f);
    invn[i*2 + 1] = 1.f / fmaxf(sqrtf(s6), 1e-12f);
}

__global__ void k_bdry(const float* __restrict__ f4, const float* __restrict__ f6,
                       const float* __restrict__ invn,
                       float* __restrict__ bh, float* __restrict__ bv) {
    int i = blockIdx.x * blockDim.x + threadIdx.x;
    if (i >= 8192) return;
    int p = i & 4095; int x = p & 63; int y = (p >> 6) & 63;
    float bhv = 0.f, bvv = 0.f;
    if (x < 63) {
        int j = i + 1; float d4 = 0.f, d6 = 0.f;
        #pragma unroll
        for (int c = 0; c < 9; ++c)  d4 += f4[i*9 + c]  * f4[j*9 + c];
        #pragma unroll
        for (int c = 0; c < 13; ++c) d6 += f6[i*13 + c] * f6[j*13 + c];
        float sim = 0.5f * (d4*invn[i*2]*invn[j*2] + d6*invn[i*2+1]*invn[j*2+1]);
        bhv = (1.f - sim) * 0.5f;
    }
    if (y < 63) {
        int j = i + 64; float d4 = 0.f, d6 = 0.f;
        #pragma unroll
        for (int c = 0; c < 9; ++c)  d4 += f4[i*9 + c]  * f4[j*9 + c];
        #pragma unroll
        for (int c = 0; c < 13; ++c) d6 += f6[i*13 + c] * f6[j*13 + c];
        float sim = 0.5f * (d4*invn[i*2]*invn[j*2] + d6*invn[i*2+1]*invn[j*2+1]);
        bvv = (1.f - sim) * 0.5f;
    }
    bh[i] = bhv; bv[i] = bvv;
}

__global__ void k_bmap(const float* __restrict__ bh, const float* __restrict__ bv,
                       float* __restrict__ bm) {
    int i = blockIdx.x * blockDim.x + threadIdx.x;
    if (i >= 8192) return;
    int p = i & 4095; int x = p & 63; int y = (p >> 6) & 63;
    float s = 0.f, cnt = 0.f;
    if (x < 63) { s += bh[i];      cnt += 1.f; }
    if (x > 0)  { s += bh[i - 1];  cnt += 1.f; }
    if (y < 63) { s += bv[i];      cnt += 1.f; }
    if (y > 0)  { s += bv[i - 64]; cnt += 1.f; }
    bm[i] = s / fmaxf(cnt, 1.f);
}

// ------------------------------- main kernel -------------------------------
// 64 threads = 2 wave32; block = one 4x4 LR block x 4 subpixel phases.
// Wave w handles two 16-pixel M-tiles (subpixel phase shared per tile).
__global__ __launch_bounds__(64)
void k_main(const float* __restrict__ f4, const float* __restrict__ f6,
            const float* __restrict__ lsig, const float* __restrict__ llam,
            const float* __restrict__ lgam,
            const _Float16* __restrict__ gmat, const _Float16* __restrict__ cgo2,
            const float* __restrict__ invn, const float* __restrict__ bh,
            const float* __restrict__ bv, const float* __restrict__ bm,
            float* __restrict__ out) {
    __shared__ __attribute__((aligned(16))) _Float16 sZ[64*224]; // attn-scaled window vec, padded
    __shared__ __attribute__((aligned(16))) _Float16 sCtx[64*32];// context, padded 32 cols
    __shared__ float sFeat[36*22];
    __shared__ float sInv[36*2];
    __shared__ float sBR[16], sBL[16], sBD[16], sBU[16], sBC[16];
    __shared__ float sBase[64*22];
    __shared__ float sOut[64*22];
    __shared__ float sY[2][16][16];

    const int tid  = threadIdx.x;
    const int wgid = blockIdx.x;
    const int b    = wgid >> 10;
    const int rem  = wgid & 1023;
    const int blk  = rem >> 2;
    const int quarter = rem & 3;          // 4 subpixel phases per block
    const int by = blk >> 4, bx = blk & 15;

    // ---- stage 0: cooperative LDS fill ----
    for (int i = tid; i < 36*22; i += 64) {
        int pt = i / 22, c = i - pt*22;
        int pr = pt / 6, pc = pt - pr*6;
        int iy = iclampi(by*4 - 1 + pr, 0, 63);
        int ix = iclampi(bx*4 - 1 + pc, 0, 63);
        int lr = b*4096 + iy*64 + ix;
        sFeat[i] = (c < 9) ? f4[lr*9 + c] : f6[lr*13 + (c - 9)];
    }
    if (tid < 36) {
        int pr = tid / 6, pc = tid - pr*6;
        int iy = iclampi(by*4 - 1 + pr, 0, 63);
        int ix = iclampi(bx*4 - 1 + pc, 0, 63);
        int lr = b*4096 + iy*64 + ix;
        sInv[tid*2 + 0] = invn[lr*2 + 0];
        sInv[tid*2 + 1] = invn[lr*2 + 1];
    }
    if (tid < 16) {
        int ly = tid >> 2, lx = tid & 3;
        int iy = by*4 + ly, ix = bx*4 + lx;
        int g = b*4096 + iy*64 + ix;
        sBR[tid] = (ix < 63) ? bh[g]      : 0.f;
        sBL[tid] = (ix > 0)  ? bh[g - 1]  : 0.f;
        sBD[tid] = (iy < 63) ? bv[g]      : 0.f;
        sBU[tid] = (iy > 0)  ? bv[g - 64] : 0.f;
        sBC[tid] = bm[g];
    }
    for (int i = tid; i < 64*22; i += 64) sOut[i] = 0.f;
    __syncthreads();

    const int lane = tid & 31, wv = tid >> 5;
    const int pp = tid;
    const int sloc = pp >> 4, m = pp & 15;
    const int s  = quarter*4 + sloc;
    const int sy = s >> 2, sx = s & 3;
    const int ly = m >> 2, lx = m & 3;
    const int pcy = 1 + ly, pcx = 1 + lx;

    // ---- stage 1: attention (registers), z staging, blended base ----
    {
        float sig = fminf(expf(lsig[0]), 0.75f);
        float inv2s2 = 1.f / (2.f * sig * sig);
        float lam = expf(llam[0]);
        float gam = expf(lgam[0]);
        float i4c = sInv[(pcy*6 + pcx)*2 + 0];
        float i6c = sInv[(pcy*6 + pcx)*2 + 1];
        const float* fc = &sFeat[(pcy*6 + pcx)*22];
        float att[9]; float mx = -1e30f;
        #pragma unroll
        for (int k = 0; k < 9; ++k) {
            int oy = k/3 - 1, ox = k%3 - 1;
            int nIdx = (pcy + oy)*6 + (pcx + ox);
            const float* fn = &sFeat[nIdx*22];
            float d4 = 0.f, d6 = 0.f;
            #pragma unroll
            for (int c = 0; c < 9; ++c)  d4 += fc[c]*fn[c];
            #pragma unroll
            for (int c = 9; c < 22; ++c) d6 += fc[c]*fn[c];
            float sim = 0.5f*(d4*i4c*sInv[nIdx*2] + d6*i6c*sInv[nIdx*2 + 1]);
            float dy = sy*0.25f - (float)oy;
            float dx = sx*0.25f - (float)ox;
            float gx = (ox > 0) ? sBR[m] : ((ox < 0) ? sBL[m] : 0.f);
            float gy = (oy > 0) ? sBD[m] : ((oy < 0) ? sBU[m] : 0.f);
            att[k] = -(dx*dx + dy*dy)*inv2s2 - lam*fmaxf(gx, gy) + gam*sim;
            mx = fmaxf(mx, att[k]);
        }
        float sum = 0.f;
        #pragma unroll
        for (int k = 0; k < 9; ++k) { att[k] = expf(att[k] - mx); sum += att[k]; }
        float isum = 1.f / sum;

        // z[pixel][k*22+c] = attn_k * window feature (padded to 224 halfs)
        unsigned* zw = (unsigned*)(sZ + pp*224);
        #pragma unroll
        for (int k = 0; k < 9; ++k) {
            int oy = k/3 - 1, ox = k%3 - 1;
            const float* fn = &sFeat[((pcy + oy)*6 + (pcx + ox))*22];
            float a = att[k]*isum;
            #pragma unroll
            for (int d = 0; d < 11; ++d)
                zw[k*11 + d] = pkh(a*fn[2*d], a*fn[2*d + 1]);
        }
        #pragma unroll
        for (int d = 99; d < 112; ++d) zw[d] = 0u;

        // bilinear base (align_corners=False), then boundary blend
        int yo = (sy >= 2) ? 1 : 0, xo = (sx >= 2) ? 1 : 0;
        int pr0 = ly + yo, pc0 = lx + xo;
        float wy = sy*0.25f - 0.375f + (yo ? 0.f : 1.f);
        float wx = sx*0.25f - 0.375f + (xo ? 0.f : 1.f);
        float tb = 1.f - sBC[m]; tb = tb*tb;
        const float* f00 = &sFeat[(pr0*6 + pc0)*22];
        const float* f01 = &sFeat[(pr0*6 + pc0 + 1)*22];
        const float* f10 = &sFeat[((pr0 + 1)*6 + pc0)*22];
        const float* f11 = &sFeat[((pr0 + 1)*6 + pc0 + 1)*22];
        #pragma unroll
        for (int c = 0; c < 22; ++c) {
            float r0 = f00[c]*(1.f - wx) + f01[c]*wx;
            float r1 = f10[c]*(1.f - wx) + f11[c]*wx;
            float bil = r0*(1.f - wy) + r1*wy;
            sBase[pp*22 + c] = tb*bil + (1.f - tb)*fc[c];
        }
    }
    __syncthreads();

    // ---- stage 2: context = G_rpos · z via WMMA (A = 2x ds_load_b128) ----
    const int mA = lane & 15;
    const int hi = (lane >> 4) & 1;
    #pragma unroll
    for (int t = 0; t < 2; ++t) {
        const int sT = quarter*4 + 2*wv + t;    // subpixel phase of this tile
        const int tileBase = (2*wv + t)*16;
        const int pA = tileBase + mA;
        const _Float16* zrow  = sZ + pA*224 + hi*16;
        const _Float16* gbase = gmat + (size_t)sT*7168 + (size_t)lane*16;
        v8f acc0 = {0.f,0.f,0.f,0.f,0.f,0.f,0.f,0.f};
        v8f acc1 = {0.f,0.f,0.f,0.f,0.f,0.f,0.f,0.f};
        for (int ks = 0; ks < 7; ++ks) {
            v8h zl = *(const v8h*)(zrow + ks*32);
            v8h zh = *(const v8h*)(zrow + ks*32 + 8);
            v16h av = __builtin_shufflevector(zl, zh, 0,1,2,3,4,5,6,7,8,9,10,11,12,13,14,15);
            v16h b0 = *(const v16h*)(gbase + ks*1024);
            v16h b1 = *(const v16h*)(gbase + ks*1024 + 512);
            acc0 = __builtin_amdgcn_wmma_f32_16x16x32_f16(false, av, false, b0, (short)0, acc0, false, false);
            acc1 = __builtin_amdgcn_wmma_f32_16x16x32_f16(false, av, false, b1, (short)0, acc1, false, false);
        }
        // C layout: M = vgpr + 8*hi, N = lane%16; pad cols (>=22) are exact 0
        #pragma unroll
        for (int vv = 0; vv < 8; ++vv) {
            int M = vv + (hi ? 8 : 0);
            sCtx[(tileBase + M)*32 + mA]      = (_Float16)acc0[vv];
            sCtx[(tileBase + M)*32 + 16 + mA] = (_Float16)acc1[vv];
        }
    }
    __syncthreads();

    // ---- stage 3: out TP: Y = ctx · CGo2 (WMMA), then rank-1 combine ----
    for (int t = 0; t < 2; ++t) {
        const int tileBase = (2*wv + t)*16;
        const int pA = tileBase + mA;
        const _Float16* crow = sCtx + pA*32 + hi*16;
        v8h cl = *(const v8h*)(crow);
        v8h ch = *(const v8h*)(crow + 8);
        v16h av = __builtin_shufflevector(cl, ch, 0,1,2,3,4,5,6,7,8,9,10,11,12,13,14,15);
        const bool owner = ((lane >> 4) == t);
        for (int q = 0; q < 31; ++q) {
            v16h bq = *(const v16h*)(cgo2 + (size_t)(q*32 + lane)*16);
            v8f acc = {0.f,0.f,0.f,0.f,0.f,0.f,0.f,0.f};
            acc = __builtin_amdgcn_wmma_f32_16x16x32_f16(false, av, false, bq, (short)0, acc, false, false);
            #pragma unroll
            for (int vv = 0; vv < 8; ++vv)
                sY[wv][vv + (hi ? 8 : 0)][mA] = acc[vv];
            __builtin_amdgcn_wave_barrier();
            if (owner) {                         // lanes owning this tile's pixels
                #pragma unroll
                for (int cc = 0; cc < 16; ++cc) {
                    int col = q*16 + cc;         // col = c1*22 + c3
                    int c1 = col / 22;  c1 = c1 > 21 ? 21 : c1;
                    int c3 = col - c1*22; c3 = c3 > 21 ? 21 : c3;
                    float w = (col < 484) ? sBase[pp*22 + c1] * sY[wv][m][cc] : 0.f;
                    sOut[pp*22 + c3] += w;
                }
            }
            __builtin_amdgcn_wave_barrier();
        }
    }

    // ---- stage 4: out = TP + base (own-thread data only) ----
    {
        const int yh = ((by*4 + ly) << 2) + sy;
        const int xh = ((bx*4 + lx) << 2) + sx;
        float* o = out + ((size_t)b*65536 + (size_t)yh*256 + xh)*22;
        #pragma unroll
        for (int c = 0; c < 22; ++c) o[c] = sOut[pp*22 + c] + sBase[pp*22 + c];
    }
}

// ------------------------------- launcher ----------------------------------
extern "C" void kernel_launch(void* const* d_in, const int* in_sizes, int n_in,
                              void* d_out, int out_size, void* d_ws, size_t ws_size,
                              hipStream_t stream) {
    (void)in_sizes; (void)n_in; (void)out_size; (void)ws_size;
    const float* f4   = (const float*)d_in[0];
    const float* f6   = (const float*)d_in[1];
    const float* lsig = (const float*)d_in[2];
    const float* llam = (const float*)d_in[3];
    const float* lgam = (const float*)d_in[4];
    const float* wval = (const float*)d_in[5];
    const float* wout = (const float*)d_in[6];
    char* ws = (char*)d_ws;                       // needs ~708 KB scratch
    float*    cg   = (float*)   (ws + WS_CG);
    _Float16* gmat = (_Float16*)(ws + WS_GMAT);
    _Float16* go   = (_Float16*)(ws + WS_CGO2);
    float*    invn = (float*)   (ws + WS_INV);
    float*    bhp  = (float*)   (ws + WS_BH);
    float*    bvp  = (float*)   (ws + WS_BV);
    float*    bmp  = (float*)   (ws + WS_BM);
    float*    o    = (float*)d_out;

    k_cg  <<<dim3(1),   dim3(32),  0, stream>>>(cg);
    k_gmat<<<dim3(896), dim3(256), 0, stream>>>(cg, wval, gmat);
    k_cgo2<<<dim3(62),  dim3(256), 0, stream>>>(cg, wout, go);
    k_norm<<<dim3(32),  dim3(256), 0, stream>>>(f4, f6, invn);
    k_bdry<<<dim3(32),  dim3(256), 0, stream>>>(f4, f6, invn, bhp, bvp);
    k_bmap<<<dim3(32),  dim3(256), 0, stream>>>(bhp, bvp, bmp);
    k_main<<<dim3(2048), dim3(64), 0, stream>>>(f4, f6, lsig, llam, lgam,
                                                gmat, go, invn, bhp, bvp, bmp, o);
}
